// PositionNet_17849884082269
// MI455X (gfx1250) — matrix operand order repaired and checked by reading, not compiled, not run on hardware
//
#include <hip/hip_runtime.h>

typedef __attribute__((ext_vector_type(16))) __bf16        v16bf;
typedef __attribute__((ext_vector_type(8)))  float         v8f;
typedef __attribute__((ext_vector_type(8)))  unsigned int  v8u;
typedef __attribute__((ext_vector_type(2)))  int           v2i;

#define JOINTS    25
#define OUTCH     200
#define MT        13          // M tiles of 16 -> 208 padded rows
#define KCH       24          // K chunks of 32 -> 768
#define FEAT      768
#define HW        64
#define HM_STRIDE 68          // padded LDS row stride in dwords (16B aligned, bank-shifted)

#if defined(__has_builtin)
#if __has_builtin(__builtin_amdgcn_global_load_async_to_lds_b64)
#define HAVE_ASYNC_LDS 1
#endif
#endif

// ---------------------------------------------------------------------------
// Repack conv_w [200,768] f32 row-major -> bf16 pairs in WMMA A-fragment order.
// Fragment (mt, kc): 32 lanes x 8 dwords. ISA 16-bit A 16x32 layout:
//   lanes 0-15 : dword j<4 -> K = 2j,2j+1 ;  j>=4 -> K = 16+2(j-4), +1
//   lanes 16-31: same +8 on K.  Row M = mt*16 + (lane&15).
// ---------------------------------------------------------------------------
__global__ void pack_w_kernel(const float* __restrict__ conv_w,
                              unsigned int* __restrict__ wpk) {
  int idx = blockIdx.x * 256 + threadIdx.x;          // MT*KCH*32*8 = 79872
  if (idx >= MT * KCH * 32 * 8) return;
  int j    = idx & 7;
  int lane = (idx >> 3) & 31;
  int blk  = idx >> 8;                               // mt*KCH + kc
  int kc   = blk % KCH;
  int mt   = blk / KCH;
  int m    = mt * 16 + (lane & 15);
  int kin  = (j < 4) ? (2 * j) : (16 + 2 * (j - 4));
  kin += (lane >> 4) * 8;
  int k = kc * 32 + kin;
  unsigned int packed = 0u;
  if (m < OUTCH) {
    __bf16 b0 = (__bf16)conv_w[m * FEAT + k];
    __bf16 b1 = (__bf16)conv_w[m * FEAT + k + 1];
    unsigned short u0 = __builtin_bit_cast(unsigned short, b0);
    unsigned short u1 = __builtin_bit_cast(unsigned short, b1);
    packed = (unsigned int)u0 | ((unsigned int)u1 << 16);
  }
  wpk[idx] = packed;
}

// ---------------------------------------------------------------------------
// Fused: per-image GEMM (bf16 WMMA, f32 accum) + bias + softmax + soft-argmax.
// One block per image. 4 waves; wave w owns pixels [w*16, w*16+16).
// W chunks double-buffered in LDS via async global->LDS copies (ASYNCcnt).
// ---------------------------------------------------------------------------
__global__ __launch_bounds__(128) void posnet_kernel(
    const float* __restrict__ feat,        // [B,768,8,8]
    const float* __restrict__ bias,        // [200]
    const unsigned int* __restrict__ wpk,  // packed W fragments
    float* __restrict__ out, int B) {
  __shared__ float smem[208 * HM_STRIDE + 208];
  float* hmLDS  = smem;                        // 208*68 floats (post-GEMM phase)
  float* biasLD = smem + 208 * HM_STRIDE;
  // K-loop phase: two 3328-dword W-chunk buffers overlay the hm region
  unsigned int* wbuf0 = (unsigned int*)smem;
  unsigned int* wbuf1 = wbuf0 + MT * 256;

  const int tid  = threadIdx.x;
  const int wave = tid >> 5;
  const int lane = tid & 31;
  const int hi   = lane >> 4;
  const int pix  = wave * 16 + (lane & 15);
  const int b    = blockIdx.x;
  const size_t bOff = (size_t)b * FEAT * HW;

  for (int i = tid; i < 208; i += 128) biasLD[i] = (i < OUTCH) ? bias[i] : 0.0f;

  // ---- W-chunk staging: 1664 qwords per chunk, 13 per thread ---------------
  const unsigned long long* wpkq = (const unsigned long long*)wpk;
#if HAVE_ASYNC_LDS
  typedef __attribute__((address_space(1))) v2i* gptr_b64;
  typedef __attribute__((address_space(3))) v2i* lptr_b64;
#define STAGE_CHUNK(kc_, dst_)                                                 \
  do {                                                                         \
    unsigned long long* dq_ = (unsigned long long*)(dst_);                     \
    _Pragma("unroll")                                                          \
    for (int t_ = 0; t_ < 13; ++t_) {                                          \
      int q_  = tid + t_ * 128;            /* qword in chunk */                \
      int mt_ = q_ >> 7;                                                       \
      size_t gq_ = ((size_t)(mt_ * KCH + (kc_)) << 7) + (q_ & 127);            \
      __builtin_amdgcn_global_load_async_to_lds_b64(                           \
          (gptr_b64)(wpkq + gq_), (lptr_b64)(dq_ + q_), 0, 0);                 \
    }                                                                          \
  } while (0)
#define WAIT_STAGE()                                                           \
  do {                                                                         \
    __builtin_amdgcn_s_wait_asynccnt(0);                                       \
  } while (0)
#else
#define STAGE_CHUNK(kc_, dst_)                                                 \
  do {                                                                         \
    unsigned long long* dq_ = (unsigned long long*)(dst_);                     \
    _Pragma("unroll")                                                          \
    for (int t_ = 0; t_ < 13; ++t_) {                                          \
      int q_  = tid + t_ * 128;                                                \
      int mt_ = q_ >> 7;                                                       \
      size_t gq_ = ((size_t)(mt_ * KCH + (kc_)) << 7) + (q_ & 127);            \
      dq_[q_] = wpkq[gq_];                                                     \
    }                                                                          \
  } while (0)
#define WAIT_STAGE() do { } while (0)
#endif

  v8f acc[MT] = {};

  STAGE_CHUNK(0, wbuf0);

  for (int kc = 0; kc < KCH; ++kc) {
    const unsigned int* cur = (kc & 1) ? wbuf1 : wbuf0;
    unsigned int* nxt = (kc & 1) ? wbuf0 : wbuf1;
    WAIT_STAGE();            // chunk kc resident in LDS (this wave's portion)
    __syncthreads();         // all waves staged kc; all reads of nxt finished
    if (kc + 1 < KCH) STAGE_CHUNK(kc + 1, nxt);   // drain behind the WMMAs

    // B fragment (ISA 16-bit B 32x16): lanes 0-15 K=0..7,16..23, lanes 16-31 K=8..15,24..31
    v16bf bfrag;
    const int cbase = kc * 32 + hi * 8;
#pragma unroll
    for (int j = 0; j < 8; ++j) {
      float f0 = feat[bOff + (size_t)(cbase + 2 * j + (j >> 2) * 8) * HW + pix];
      float f1 = feat[bOff + (size_t)(cbase + 2 * j + (j >> 2) * 8 + 1) * HW + pix];
      bfrag[2 * j]     = (__bf16)f0;
      bfrag[2 * j + 1] = (__bf16)f1;
    }

    // A fragments pipelined one tile ahead so ds_load hides under WMMA
    const unsigned int* cw = cur + lane * 8;
    v8u au = *(const v8u*)cw;
#pragma unroll
    for (int mt = 0; mt < MT; ++mt) {
      v8u aun = au;
      if (mt + 1 < MT) aun = *(const v8u*)(cw + (mt + 1) * 256);
      v16bf afrag = __builtin_bit_cast(v16bf, au);
      acc[mt] = __builtin_amdgcn_wmma_f32_16x16x32_bf16(
          false, afrag, false, bfrag, (short)0, acc[mt], false, false);
      au = aun;
    }
  }

  __syncthreads();
  // C/D layout: lanes 0-15: M = mt*16+v, N=lane ; lanes 16-31: M = mt*16+8+v
#pragma unroll
  for (int mt = 0; mt < MT; ++mt) {
#pragma unroll
    for (int v = 0; v < 8; ++v) {
      int m = mt * 16 + hi * 8 + v;
      hmLDS[m * HM_STRIDE + pix] = acc[mt][v] + biasLD[m];
    }
  }
  __syncthreads();

  // ---- softmax over 512 logits per (b, joint) + soft-argmax ----------------
  float* outHM = out;
  float* outC  = out + (size_t)B * JOINTS * 512;
  for (int j = wave; j < JOINTS; j += 4) {
    float pv[16];
    float mx = -3.0e38f;
#pragma unroll
    for (int t = 0; t < 4; ++t) {
      int f = 4 * (lane + 32 * t);                     // flat d*64+h*8+w
      int o = j * 8 + (f >> 6);
      const float4 q = *(const float4*)&hmLDS[o * HM_STRIDE + (f & 63)];
      pv[4 * t + 0] = q.x; pv[4 * t + 1] = q.y;
      pv[4 * t + 2] = q.z; pv[4 * t + 3] = q.w;
      mx = fmaxf(mx, fmaxf(fmaxf(q.x, q.y), fmaxf(q.z, q.w)));
    }
    for (int off = 16; off > 0; off >>= 1) mx = fmaxf(mx, __shfl_xor(mx, off, 32));
    float sum = 0.0f;
#pragma unroll
    for (int i = 0; i < 16; ++i) { pv[i] = __expf(pv[i] - mx); sum += pv[i]; }
    for (int off = 16; off > 0; off >>= 1) sum += __shfl_xor(sum, off, 32);
    const float inv = 1.0f / sum;

    float sx = 0.f, sy = 0.f, sz = 0.f;
#pragma unroll
    for (int t = 0; t < 4; ++t) {
      int f = 4 * (lane + 32 * t);
      float4 q;
      q.x = pv[4 * t + 0] * inv; q.y = pv[4 * t + 1] * inv;
      q.z = pv[4 * t + 2] * inv; q.w = pv[4 * t + 3] * inv;
      float fd = (float)(f >> 6);
#pragma unroll
      for (int e = 0; e < 4; ++e) {
        float p = (&q.x)[e];
        int fe = f + e;
        sx += p * (float)(fe & 7);
        sy += p * (float)((fe >> 3) & 7);
        sz += p * fd;
      }
      *(float4*)&outHM[(size_t)b * (JOINTS * 512) + j * 512 + f] = q;
    }
    for (int off = 16; off > 0; off >>= 1) {
      sx += __shfl_xor(sx, off, 32);
      sy += __shfl_xor(sy, off, 32);
      sz += __shfl_xor(sz, off, 32);
    }
    if (lane == 0) {
      size_t cb = ((size_t)b * JOINTS + j) * 3;
      outC[cb + 0] = sx; outC[cb + 1] = sy; outC[cb + 2] = sz;
    }
  }
}

extern "C" void kernel_launch(void* const* d_in, const int* in_sizes, int n_in,
                              void* d_out, int out_size, void* d_ws, size_t ws_size,
                              hipStream_t stream) {
  const float* img_feat = (const float*)d_in[0];
  const float* conv_w   = (const float*)d_in[1];
  const float* conv_b   = (const float*)d_in[2];
  float* outp           = (float*)d_out;
  unsigned int* wpk     = (unsigned int*)d_ws;   // needs 79872*4 = 319,488 B

  const int B = in_sizes[0] / (FEAT * HW);       // 512

  pack_w_kernel<<<(MT * KCH * 32 * 8 + 255) / 256, 256, 0, stream>>>(conv_w, wpk);
  posnet_kernel<<<B, 128, 0, stream>>>(img_feat, conv_b, wpk, outp, B);
}
